// LSTM_GNN_54288386621669
// MI455X (gfx1250) — matrix operand order repaired and checked by reading, not compiled, not run on hardware
//
#include <hip/hip_runtime.h>
#include <hip/hip_bf16.h>

// ---------------------------------------------------------------------------
// GCN(64->128) -> GCN(128->128) -> LSTM(128->256, T=32) -> Linear(256->1)
// GEMMs on CDNA5 WMMA (v_wmma_f32_16x16x32_bf16), f32 accumulate.
//   A operand: staged per-block into LDS via global_load_async_to_lds_b128
//              (ASYNCcnt + s_wait_asynccnt), read back as ds_load_b128 + cvt.
//   B operand: weights staged per-block into LDS as bf16 [n][k] -> ds_load_b128.
// ---------------------------------------------------------------------------

#define T_STEPS 32
#define N_NODES 20000
#define E_EDGES 320000
#define CIN     64
#define H1C     128
#define H2C     128
#define HL      256
#define G4      (4 * HL)   // 1024 gate columns

typedef __attribute__((ext_vector_type(16))) __bf16 v16bf;
typedef __attribute__((ext_vector_type(8)))  float  v8f;

static __device__ __forceinline__ v8f wmma_bf16(v16bf a, v16bf b, v8f c) {
    // (neg_a, A, neg_b, B, c_mod, C, reuse_a, reuse_b)
    return __builtin_amdgcn_wmma_f32_16x16x32_bf16(false, a, false, b,
                                                   (short)0, c, false, false);
}

// Async 16B copy global -> LDS (CDNA5 GLOBAL_LOAD_ASYNC_TO_LDS_B128, ASYNCcnt).
// ldsPtr: flat pointer into __shared__ (low 32 bits = LDS byte address).
static __device__ __forceinline__ void async_copy_b128(void* ldsPtr, const void* gPtr) {
    unsigned loff = (unsigned)(unsigned long long)ldsPtr;
    unsigned long long ga = (unsigned long long)gPtr;
    asm volatile("global_load_async_to_lds_b128 %0, %1, off"
                 :: "v"(loff), "v"(ga) : "memory");
}
static __device__ __forceinline__ void async_wait0() {
    asm volatile("s_wait_asynccnt 0x0" ::: "memory");
}

// A fragment 16(M)x32(K): row r (local), chunk k0, lane-half h, from LDS f32
// tile with row stride `stride` floats (stride*4 % 32 == 0 -> aligned b128).
// elems 0..15 -> K = k0 + 16h + e
static __device__ __forceinline__ v16bf load_a_lds(const float* __restrict__ sA,
                                                   int stride, int r, int k0, int h) {
    const float* p = sA + r * stride + k0 + 16 * h;
    const float4 x0 = ((const float4*)p)[0];
    const float4 x1 = ((const float4*)p)[1];
    const float4 x2 = ((const float4*)p)[2];
    const float4 x3 = ((const float4*)p)[3];
    v16bf a;
    a[0] = (__bf16)x0.x;  a[1] = (__bf16)x0.y;  a[2] = (__bf16)x0.z;  a[3] = (__bf16)x0.w;
    a[4] = (__bf16)x1.x;  a[5] = (__bf16)x1.y;  a[6] = (__bf16)x1.z;  a[7] = (__bf16)x1.w;
    a[8] = (__bf16)x2.x;  a[9] = (__bf16)x2.y;  a[10] = (__bf16)x2.z; a[11] = (__bf16)x2.w;
    a[12] = (__bf16)x3.x; a[13] = (__bf16)x3.y; a[14] = (__bf16)x3.z; a[15] = (__bf16)x3.w;
    return a;
}

// B fragment from LDS staged as bf16 [nLocal][kpad] (kpad multiple of 16).
// Lane L: column nl0 + L%16, K = k0 + 16*(L/16) + e -> one aligned 32B LDS read.
static __device__ __forceinline__ v16bf load_b_lds(const __bf16* __restrict__ lds,
                                                   int kpad, int nl0, int k0, int lane) {
    const int h = (lane >> 4) & 1;
    const __bf16* p = lds + (long long)(nl0 + (lane & 15)) * kpad + k0 + 16 * h;
    return *(const v16bf*)p;   // 32B aligned: kpad%16==0, k0%32==0
}

// ---------------------------------------------------------------------------
// Utility kernels
// ---------------------------------------------------------------------------
__global__ void fill_kernel(float* __restrict__ p, long long n, float v) {
    long long i = (long long)blockIdx.x * blockDim.x + threadIdx.x;
    if (i < n) p[i] = v;
}

__global__ void degree_kernel(const long long* __restrict__ ei, float* __restrict__ deg) {
    int e = blockIdx.x * blockDim.x + threadIdx.x;
    if (e < E_EDGES) atomicAdd(&deg[(int)ei[E_EDGES + e]], 1.0f);
}

__global__ void dinv_kernel(float* __restrict__ deg) {
    int n = blockIdx.x * blockDim.x + threadIdx.x;
    if (n < N_NODES) deg[n] = rsqrtf(deg[n] + 1.0f);   // self-loop degree
}

// ---------------------------------------------------------------------------
// GCN dense transform: H = X @ W   (X [T*N, K] row-major, W [K, 128])
// Block: 256 thr = 8 waves; wave w -> cols [16w,16w+16); block -> rows [32b,32b+32).
// A rows async-staged into LDS once per block (shared by all 8 waves).
// ---------------------------------------------------------------------------
__global__ void __launch_bounds__(256)
gcn_gemm_kernel(const float* __restrict__ X, const float* __restrict__ W,
                float* __restrict__ Hout, int K) {
    __shared__ unsigned short sW[128 * (H1C + 16)];       // bf16 [128][K+16]
    __shared__ float sA[32 * (H1C + 8)];                  // f32 [32][K+8]
    __bf16* lw = (__bf16*)sW;
    const int kpad = K + 16;
    const int astr = K + 8;                               // (K+8)*4 % 32 == 0
    const int m0 = blockIdx.x * 32;

    // async-stage A tile: 32 rows x K f32, 16B chunks (8*K of them; %256==0)
    for (int f = threadIdx.x; f < 8 * K; f += 256) {
        const int row = f / (K >> 2);
        const int c4  = (f % (K >> 2)) << 2;
        async_copy_b128(&sA[row * astr + c4],
                        X + (long long)(m0 + row) * K + c4);
    }
    // transpose-stage W[k][n] -> lw[n][k] with bf16 convert (coalesced reads)
    for (int f = threadIdx.x; f < K * 128; f += 256) {
        const int k = f >> 7, nl = f & 127;
        lw[(long long)nl * kpad + k] = (__bf16)W[(long long)k * 128 + nl];
    }
    async_wait0();
    __syncthreads();

    const int lane = threadIdx.x & 31;
    const int wave = threadIdx.x >> 5;
    const int nl0 = wave * 16;
    const int h = (lane >> 4) & 1;
    const int rA = lane & 15;
    v8f acc0 = {}, acc1 = {};
    for (int k0 = 0; k0 < K; k0 += 32) {
        v16bf b  = load_b_lds(lw, kpad, nl0, k0, lane);
        v16bf a0 = load_a_lds(sA, astr, rA, k0, h);
        v16bf a1 = load_a_lds(sA, astr, 16 + rA, k0, h);
        acc0 = wmma_bf16(a0, b, acc0);
        acc1 = wmma_bf16(a1, b, acc1);
    }
    const int col = nl0 + (lane & 15);
#pragma unroll
    for (int v = 0; v < 8; ++v) {            // C/D layout: VGPR v -> M = v + 8*half
        Hout[(long long)(m0 + 8 * h + v) * 128 + col]      = acc0[v];
        Hout[(long long)(m0 + 16 + 8 * h + v) * 128 + col] = acc1[v];
    }
}

// ---------------------------------------------------------------------------
// Normalized scatter-add over edges: Agg[t,dst,c] += H[t,src,c]*dinv[s]*dinv[d]
// ---------------------------------------------------------------------------
__global__ void __launch_bounds__(128)
scatter_kernel(const long long* __restrict__ ei, const float* __restrict__ dinv,
               const float* __restrict__ Hin, float* __restrict__ Agg) {
    const int e = blockIdx.x;
    const int s = (int)ei[e];
    const int d = (int)ei[E_EDGES + e];
    const float norm = dinv[s] * dinv[d];
    const int c = threadIdx.x;
#pragma unroll 4
    for (int t = 0; t < T_STEPS; ++t) {
        long long base = (long long)t * N_NODES * 128;
        if (t + 1 < T_STEPS)   // global_prefetch_b8 next timestep's source line
            __builtin_prefetch(&Hin[base + (long long)N_NODES * 128 + (long long)s * 128 + c], 0, 0);
        atomicAdd(&Agg[base + (long long)d * 128 + c],
                  Hin[base + (long long)s * 128 + c] * norm);
    }
}

// Out = relu(Agg + H*dinv^2 + bias)
__global__ void gcn_finish_kernel(const float* __restrict__ Hin, const float* __restrict__ Agg,
                                  const float* __restrict__ dinv, const float* __restrict__ bias,
                                  float* __restrict__ Out) {
    long long i = (long long)blockIdx.x * blockDim.x + threadIdx.x;
    const long long total = (long long)T_STEPS * N_NODES * 128;
    if (i >= total) return;
    int c = (int)(i & 127);
    int n = (int)((i >> 7) % N_NODES);
    float di = dinv[n];
    float v = Agg[i] + Hin[i] * di * di + bias[c];
    Out[i] = v > 0.0f ? v : 0.0f;
}

// ---------------------------------------------------------------------------
// LSTM gates: Gates = Xt @ Wih^T + Hprev @ Whh^T + bih + bhh   [N, 1024]
// grid = (N/32, 8); block stages its 128-col slice of Wih/Whh (bf16) plus the
// 32-row Xt and Hprev A tiles (async f32) into LDS.
// ---------------------------------------------------------------------------
__global__ void __launch_bounds__(256)
lstm_gates_kernel(const float* __restrict__ Xt, const float* __restrict__ Hprev,
                  const float* __restrict__ Wih, const float* __restrict__ Whh,
                  const float* __restrict__ bih, const float* __restrict__ bhh,
                  float* __restrict__ Gates) {
    __shared__ unsigned short sIH[128 * (H2C + 16)];      // bf16, kpad 144
    __shared__ unsigned short sHH[128 * (HL + 16)];       // bf16, kpad 272
    __shared__ float sAX[32 * (H2C + 8)];                 // f32 [32][136]
    __shared__ float sAH[32 * (HL + 8)];                  // f32 [32][264]
    __bf16* lih = (__bf16*)sIH;
    __bf16* lhh = (__bf16*)sHH;
    const int nbase = blockIdx.y * 128;
    const int m0 = blockIdx.x * 32;
    const int astrX = H2C + 8, astrH = HL + 8;

    // async-stage A tiles (16B chunks; 8*128=1024 and 8*256=2048, both %256==0)
    for (int f = threadIdx.x; f < 8 * H2C; f += 256) {
        const int row = f / (H2C >> 2);
        const int c4  = (f % (H2C >> 2)) << 2;
        async_copy_b128(&sAX[row * astrX + c4],
                        Xt + (long long)(m0 + row) * H2C + c4);
    }
    for (int f = threadIdx.x; f < 8 * HL; f += 256) {
        const int row = f / (HL >> 2);
        const int c4  = (f % (HL >> 2)) << 2;
        async_copy_b128(&sAH[row * astrH + c4],
                        Hprev + (long long)(m0 + row) * HL + c4);
    }
    // weight slices: torch [out,in] already [n][k]; straight copy + bf16 cvt
    for (int f = threadIdx.x; f < 128 * H2C; f += 256) {
        const int nl = f >> 7, k = f & (H2C - 1);
        lih[(long long)nl * (H2C + 16) + k] = (__bf16)Wih[(long long)(nbase + nl) * H2C + k];
    }
    for (int f = threadIdx.x; f < 128 * HL; f += 256) {
        const int nl = f >> 8, k = f & (HL - 1);
        lhh[(long long)nl * (HL + 16) + k] = (__bf16)Whh[(long long)(nbase + nl) * HL + k];
    }
    async_wait0();
    __syncthreads();

    const int lane = threadIdx.x & 31;
    const int wave = threadIdx.x >> 5;
    const int nl0 = wave * 16;
    const int h = (lane >> 4) & 1;
    const int rA = lane & 15;
    v8f acc0 = {}, acc1 = {};
#pragma unroll
    for (int k0 = 0; k0 < H2C; k0 += 32) {     // input contribution, K = 128
        v16bf b  = load_b_lds(lih, H2C + 16, nl0, k0, lane);
        v16bf a0 = load_a_lds(sAX, astrX, rA, k0, h);
        v16bf a1 = load_a_lds(sAX, astrX, 16 + rA, k0, h);
        acc0 = wmma_bf16(a0, b, acc0);
        acc1 = wmma_bf16(a1, b, acc1);
    }
#pragma unroll
    for (int k0 = 0; k0 < HL; k0 += 32) {      // recurrent contribution, K = 256
        v16bf b  = load_b_lds(lhh, HL + 16, nl0, k0, lane);
        v16bf a0 = load_a_lds(sAH, astrH, rA, k0, h);
        v16bf a1 = load_a_lds(sAH, astrH, 16 + rA, k0, h);
        acc0 = wmma_bf16(a0, b, acc0);
        acc1 = wmma_bf16(a1, b, acc1);
    }
    const int col = nbase + nl0 + (lane & 15);
    const float bsum = bih[col] + bhh[col];
#pragma unroll
    for (int v = 0; v < 8; ++v) {
        Gates[(long long)(m0 + 8 * h + v) * G4 + col]      = acc0[v] + bsum;
        Gates[(long long)(m0 + 16 + 8 * h + v) * G4 + col] = acc1[v] + bsum;
    }
}

// Cell update: c = sig(f)*c + sig(i)*tanh(g); h = sig(o)*tanh(c)
__global__ void lstm_point_kernel(const float* __restrict__ Gates,
                                  float* __restrict__ Hs, float* __restrict__ Cs) {
    int i = blockIdx.x * blockDim.x + threadIdx.x;
    if (i >= N_NODES * HL) return;
    int n = i / HL, hh = i % HL;
    const float* g = Gates + (long long)n * G4;
    float ig = 1.0f / (1.0f + __expf(-g[hh]));
    float fg = 1.0f / (1.0f + __expf(-g[HL + hh]));
    float gg = tanhf(g[2 * HL + hh]);
    float og = 1.0f / (1.0f + __expf(-g[3 * HL + hh]));
    float c = fg * Cs[i] + ig * gg;
    Cs[i] = c;
    Hs[i] = og * tanhf(c);
}

// out[n] = hT[n,:] . Wf + bf
__global__ void final_kernel(const float* __restrict__ Hs, const float* __restrict__ Wf,
                             const float* __restrict__ bf, float* __restrict__ Out) {
    int n = blockIdx.x * blockDim.x + threadIdx.x;
    if (n >= N_NODES) return;
    const float* h = Hs + (long long)n * HL;
    float acc = 0.0f;
#pragma unroll 8
    for (int k = 0; k < HL; ++k) acc += h[k] * Wf[k];
    Out[n] = acc + bf[0];
}

// ---------------------------------------------------------------------------
extern "C" void kernel_launch(void* const* d_in, const int* in_sizes, int n_in,
                              void* d_out, int out_size, void* d_ws, size_t ws_size,
                              hipStream_t stream) {
    const float*     x   = (const float*)d_in[0];
    const long long* ei  = (const long long*)d_in[1];   // int64 [2, E]
    const float*     W1  = (const float*)d_in[2];
    const float*     b1  = (const float*)d_in[3];
    const float*     W2  = (const float*)d_in[4];
    const float*     b2  = (const float*)d_in[5];
    const float*     Wih = (const float*)d_in[6];
    const float*     Whh = (const float*)d_in[7];
    const float*     bih = (const float*)d_in[8];
    const float*     bhh = (const float*)d_in[9];
    const float*     Wf  = (const float*)d_in[10];
    const float*     bf  = (const float*)d_in[11];
    float*           out = (float*)d_out;

    // Workspace layout (floats): dinv | bufA | bufB | bufC | gates | h | c
    float* ws = (float*)d_ws;
    const long long big = (long long)T_STEPS * N_NODES * H1C;   // 81,920,000 f32
    float* dinv   = ws;
    float* bufA   = ws + 20480;
    float* bufB   = bufA + big;
    float* bufC   = bufB + big;
    float* gates  = bufC + big;
    float* hstate = gates + (long long)N_NODES * G4;
    float* cstate = hstate + (long long)N_NODES * HL;

    const long long totalTN = (long long)T_STEPS * N_NODES;     // 640,000 rows
    const unsigned bigBlocks = (unsigned)((big + 255) / 256);

    // Degrees (with self loop) -> dinv, in place
    fill_kernel<<<(20480 + 255) / 256, 256, 0, stream>>>(dinv, 20480, 0.0f);
    degree_kernel<<<(E_EDGES + 255) / 256, 256, 0, stream>>>(ei, dinv);
    dinv_kernel<<<(N_NODES + 255) / 256, 256, 0, stream>>>(dinv);

    // ---- GCN layer 1: x [T*N,64] @ W1 [64,128] ----
    gcn_gemm_kernel<<<(unsigned)(totalTN / 32), 256, 0, stream>>>(x, W1, bufA, CIN);
    fill_kernel<<<bigBlocks, 256, 0, stream>>>(bufB, big, 0.0f);
    scatter_kernel<<<E_EDGES, 128, 0, stream>>>(ei, dinv, bufA, bufB);
    gcn_finish_kernel<<<bigBlocks, 256, 0, stream>>>(bufA, bufB, dinv, b1, bufC);

    // ---- GCN layer 2: bufC [T*N,128] @ W2 [128,128] ----
    gcn_gemm_kernel<<<(unsigned)(totalTN / 32), 256, 0, stream>>>(bufC, W2, bufA, H1C);
    fill_kernel<<<bigBlocks, 256, 0, stream>>>(bufB, big, 0.0f);
    scatter_kernel<<<E_EDGES, 128, 0, stream>>>(ei, dinv, bufA, bufB);
    gcn_finish_kernel<<<bigBlocks, 256, 0, stream>>>(bufA, bufB, dinv, b2, bufC);

    // ---- LSTM over T steps, batch = nodes ----
    const long long nh = (long long)N_NODES * HL;
    fill_kernel<<<(unsigned)((nh + 255) / 256), 256, 0, stream>>>(hstate, nh, 0.0f);
    fill_kernel<<<(unsigned)((nh + 255) / 256), 256, 0, stream>>>(cstate, nh, 0.0f);
    for (int t = 0; t < T_STEPS; ++t) {
        const float* xt = bufC + (long long)t * N_NODES * H2C;
        lstm_gates_kernel<<<dim3(N_NODES / 32, 8), 256, 0, stream>>>(
            xt, hstate, Wih, Whh, bih, bhh, gates);
        lstm_point_kernel<<<(unsigned)((nh + 255) / 256), 256, 0, stream>>>(
            gates, hstate, cstate);
    }

    // ---- Final linear [N,256] -> [N,1] ----
    final_kernel<<<(N_NODES + 255) / 256, 256, 0, stream>>>(hstate, Wf, bf, out);
}
